// GCN_70901320122855
// MI455X (gfx1250) — matrix operand order repaired and checked by reading, not compiled, not run on hardware
//
#include <hip/hip_runtime.h>

// ---------------------------------------------------------------------------
// GCN 2-layer forward for MI455X (gfx1250, wave32, WMMA).
// Dense GEMMs via v_wmma_f32_16x16x32_bf16 (bf16 A/B, fp32 accumulate).
// Sparse aggregation via hardware global_atomic_add_f32 (edge-parallel).
// ---------------------------------------------------------------------------

typedef __attribute__((ext_vector_type(16))) __bf16 v16bf;
typedef __attribute__((ext_vector_type(8)))  float  v8f;

#define NNODES 50000
#define KDIM   128      // inner dim of both GEMMs (IN_F == HID_F == 128)
#define HIDF   128
#define OUTF   64

union FragB16 {
    unsigned int u[8];
    v16bf        v;
};

// --------------------------- fp32 -> bf16 copy -----------------------------
__global__ void f32_to_bf16_kernel(const float* __restrict__ in,
                                   __bf16* __restrict__ out, int n) {
    int i = blockIdx.x * blockDim.x + threadIdx.x;
    if (i < n) out[i] = (__bf16)in[i];
}

// ---- pack W[K][Nc] fp32 row-major into u32 pairs: lo = bf16(W[2kp][n]),
// ---- hi = bf16(W[2kp+1][n]) so a B-fragment VGPR is one 32-bit load.
__global__ void pack_w_kernel(const float* __restrict__ W,
                              unsigned int* __restrict__ Wp, int K, int Nc) {
    int i = blockIdx.x * blockDim.x + threadIdx.x;
    int total = (K / 2) * Nc;
    if (i >= total) return;
    int kp = i / Nc;
    int n  = i % Nc;
    __bf16 lo = (__bf16)W[(2 * kp)     * Nc + n];
    __bf16 hi = (__bf16)W[(2 * kp + 1) * Nc + n];
    unsigned int ulo = (unsigned int)__builtin_bit_cast(unsigned short, lo);
    unsigned int uhi = (unsigned int)__builtin_bit_cast(unsigned short, hi);
    Wp[i] = (uhi << 16) | ulo;
}

// --------------------------- WMMA GEMM -------------------------------------
// C[M][NC] = A[M][128](bf16) * B[128][NC] (B pre-packed as u32 K-pairs).
// One wave32 computes one 16x16 output tile; K loop = 4 x v_wmma 16x16x32.
template <int NC>
__global__ __launch_bounds__(256)
void gemm_bf16_wmma_kernel(const __bf16* __restrict__ A,
                           const unsigned int* __restrict__ Bp,
                           float* __restrict__ C, int M) {
    const int lane   = threadIdx.x & 31;
    const int wave   = threadIdx.x >> 5;
    const int tilesN = NC / 16;
    const int tilesM = M / 16;
    const int w      = blockIdx.x * (blockDim.x >> 5) + wave;
    if (w >= tilesM * tilesN) return;      // wave-uniform guard (EXEC stays full)

    const int tm   = w / tilesN;
    const int tn   = w % tilesN;
    const int half = lane >> 4;            // 0: lanes 0-15, 1: lanes 16-31
    const int l16  = lane & 15;

    // A rows as u32 (two consecutive bf16 K-values per dword, even-K in [15:0])
    const unsigned int* A32  = (const unsigned int*)A;
    const int           arow = (tm * 16 + l16) * (KDIM / 2);
    const int           bn   = tn * 16 + l16;

    v8f c = {};
#pragma unroll
    for (int k0 = 0; k0 < KDIM; k0 += 32) {
        FragB16 a, b;
        // A 16x32 bf16 fragment: VGPR v<4 -> K = k0 + 8*half + 2v{,+1}
        //                        VGPR v>=4 -> K = k0 + 16 + 8*half + 2(v-4){,+1}
        const int abase = arow + (k0 >> 1) + 4 * half;
#pragma unroll
        for (int v = 0; v < 4; ++v) a.u[v]     = A32[abase + v];
#pragma unroll
        for (int v = 0; v < 4; ++v) a.u[4 + v] = A32[abase + 8 + v];

        // B 32x16 fragment: same K order as A;
        // VGPR v -> K pair index k0/2 + 4*half + v + (v & 4), column bn.
        const int bbase = ((k0 >> 1) + 4 * half) * NC + bn;
#pragma unroll
        for (int v = 0; v < 8; ++v) b.u[v] = Bp[bbase + (v + (v & 4)) * NC];

        c = __builtin_amdgcn_wmma_f32_16x16x32_bf16(
                /*neg_a=*/false, a.v, /*neg_b=*/false, b.v,
                /*c_mod=*/(short)0, c, /*reuse_a=*/false, /*reuse_b=*/false);
    }

    // C/D 16x16 f32 layout: VGPR r -> row 8*half + r, col l16.
#pragma unroll
    for (int r = 0; r < 8; ++r) {
        const int m = tm * 16 + half * 8 + r;
        C[m * NC + tn * 16 + l16] = c[r];
    }
}

// --------------------------- edge-parallel SpMM ----------------------------
// out[row[e], :] += val[e] * sup[col[e], :]   (one wave32 per edge)
template <int F, int VEC>   // VEC = F / 32
__global__ __launch_bounds__(256)
void spmm_atomic_kernel(const int* __restrict__ row, const int* __restrict__ col,
                        const float* __restrict__ val,
                        const float* __restrict__ sup,
                        float* __restrict__ out, int E) {
    const int gwave = (int)((blockIdx.x * (unsigned)blockDim.x + threadIdx.x) >> 5);
    const int lane  = threadIdx.x & 31;
    if (gwave >= E) return;

    const int   r = row[gwave];
    const int   cidx = col[gwave];
    const float v = val[gwave];

    const float* s = sup + (long long)cidx * F + lane * VEC;
    float*       o = out + (long long)r    * F + lane * VEC;
#pragma unroll
    for (int j = 0; j < VEC; ++j) {
        unsafeAtomicAdd(&o[j], v * s[j]);   // -> global_atomic_add_f32 (no return)
    }
}

// ------------------ bias + ReLU, emit bf16 activations ---------------------
__global__ void bias_relu_bf16_kernel(const float* __restrict__ acc,
                                      const float* __restrict__ bias,
                                      __bf16* __restrict__ hb, int n, int Fmask) {
    int i = blockIdx.x * blockDim.x + threadIdx.x;
    if (i < n) {
        float t = acc[i] + bias[i & Fmask];
        hb[i] = (__bf16)(t > 0.0f ? t : 0.0f);
    }
}

// --------------------------- final bias add --------------------------------
__global__ void bias_add_kernel(float* __restrict__ out,
                                const float* __restrict__ bias, int n, int Fmask) {
    int i = blockIdx.x * blockDim.x + threadIdx.x;
    if (i < n) out[i] += bias[i & Fmask];
}

// ---------------------------------------------------------------------------
extern "C" void kernel_launch(void* const* d_in, const int* in_sizes, int n_in,
                              void* d_out, int out_size, void* d_ws, size_t ws_size,
                              hipStream_t stream) {
    const float* x    = (const float*)d_in[0];
    const int*   row0 = (const int*)d_in[1];
    const int*   col0 = (const int*)d_in[2];
    const float* val0 = (const float*)d_in[3];
    const int*   row1 = (const int*)d_in[4];
    const int*   col1 = (const int*)d_in[5];
    const float* val1 = (const float*)d_in[6];
    const float* W0   = (const float*)d_in[7];
    const float* b0   = (const float*)d_in[8];
    const float* W1   = (const float*)d_in[9];
    const float* b1   = (const float*)d_in[10];
    float* out = (float*)d_out;

    const int E = in_sizes[1];                       // 800000
    const int M = NNODES;                            // 50000 (== 16 * 3125)

    // -------- workspace partition (all 16B-aligned offsets) --------
    char* ws = (char*)d_ws;
    const size_t xb_bytes   = (size_t)M * KDIM * 2;  // 12.8 MB
    const size_t hb_bytes   = (size_t)M * HIDF * 2;  // 12.8 MB
    const size_t sup_bytes  = (size_t)M * HIDF * 4;  // 25.6 MB (reused for layer1)
    const size_t acc_bytes  = (size_t)M * HIDF * 4;  // 25.6 MB

    __bf16*       xb   = (__bf16*)(ws);
    __bf16*       hb   = (__bf16*)(ws + xb_bytes);
    float*        sup  = (float*)(ws + xb_bytes + hb_bytes);
    float*        acc0 = (float*)(ws + xb_bytes + hb_bytes + sup_bytes);
    unsigned int* W0p  = (unsigned int*)(ws + xb_bytes + hb_bytes + sup_bytes + acc_bytes);
    unsigned int* W1p  = W0p + (KDIM / 2) * HIDF;    // +32 KB

    // -------- stage 0: precision conversion / weight packing --------
    const int nX = M * KDIM;                         // 6.4M
    f32_to_bf16_kernel<<<(nX + 255) / 256, 256, 0, stream>>>(x, xb, nX);
    pack_w_kernel<<<((KDIM / 2) * HIDF + 255) / 256, 256, 0, stream>>>(W0, W0p, KDIM, HIDF);
    pack_w_kernel<<<((KDIM / 2) * OUTF + 255) / 256, 256, 0, stream>>>(W1, W1p, KDIM, OUTF);

    hipMemsetAsync(acc0, 0, acc_bytes, stream);
    hipMemsetAsync(out,  0, (size_t)out_size * sizeof(float), stream);

    // -------- layer 0 --------
    {   // sup = xb @ W0  (M x 128), one wave per 16x16 tile
        const int tiles  = (M / 16) * (HIDF / 16);   // 25000
        const int blocks = (tiles + 7) / 8;
        gemm_bf16_wmma_kernel<HIDF><<<blocks, 256, 0, stream>>>(xb, W0p, sup, M);
    }
    {   // acc0[row0] += val0 * sup[col0]
        const int blocks = (E + 7) / 8;              // 8 waves (edges) per block
        spmm_atomic_kernel<HIDF, 4><<<blocks, 256, 0, stream>>>(row0, col0, val0, sup, acc0, E);
    }
    {   // hb = bf16(relu(acc0 + b0))
        const int n = M * HIDF;
        bias_relu_bf16_kernel<<<(n + 255) / 256, 256, 0, stream>>>(acc0, b0, hb, n, HIDF - 1);
    }

    // -------- layer 1 --------
    {   // sup(first 12.8MB) = hb @ W1  (M x 64)
        const int tiles  = (M / 16) * (OUTF / 16);   // 12500
        const int blocks = (tiles + 7) / 8;
        gemm_bf16_wmma_kernel<OUTF><<<blocks, 256, 0, stream>>>(hb, W1p, sup, M);
    }
    {   // out[row1] += val1 * sup[col1]
        const int blocks = (E + 7) / 8;
        spmm_atomic_kernel<OUTF, 2><<<blocks, 256, 0, stream>>>(row1, col1, val1, sup, out, E);
    }
    {   // out += b1
        const int n = M * OUTF;
        bias_add_kernel<<<(n + 255) / 256, 256, 0, stream>>>(out, b1, n, OUTF - 1);
    }
}
